// BNN3L_86870008528978
// MI455X (gfx1250) — matrix operand order repaired and checked by reading, not compiled, not run on hardware
//
#include <hip/hip_runtime.h>
#include <hip/hip_bf16.h>

typedef __bf16 bf16;
typedef __attribute__((ext_vector_type(16))) __bf16 v16bf;
typedef __attribute__((ext_vector_type(8)))  float  v8f;

union ABf { v16bf v; uint4 q[2]; };

#define T_STEPS 1000
#define BSZ     256
#define NIN     128
#define NHID    512
#define NOUT    128
#define NWG     16
#define BLK     256          // 8 wave32s
#define HSLICE  32           // NHID / NWG
#define WREC_LD 520          // padded K-stride (elements) for LDS W_rec slice
#define WIN_LD  136          // padded K-stride for LDS W_in slice

#if defined(__has_builtin)
#if __has_builtin(__builtin_amdgcn_sched_group_barrier)
#define SGB(mask, n, id) __builtin_amdgcn_sched_group_barrier(mask, n, id)
#endif
#endif
#ifndef SGB
#define SGB(mask, n, id)
#endif
// masks: 0x008 = MFMA/WMMA, 0x020 = VMEM read, 0x100 = DS read

__device__ __forceinline__ v8f wmma_bf16(v16bf a, v16bf b, v8f c) {
  return __builtin_amdgcn_wmma_f32_16x16x32_bf16(false, a, false, b, (short)0, c,
                                                 false, false);
}

// A operand (16x32 bf16 tile) from row-major memory.
// lane<16: row, K = k0..k0+15 ; lane>=16: same row, K = k0+16..k0+31 (caller adds half*16).
__device__ __forceinline__ v16bf load_a_g(const bf16* __restrict__ base, int row,
                                          int k0, int ldk) {
  ABf r;
  const bf16* p = base + (size_t)row * ldk + k0;
  r.q[0] = *(const uint4*)(p);
  r.q[1] = *(const uint4*)(p + 8);
  return r.v;
}

// B operand (32x16 bf16 tile): per-lane column, contiguous K.
// lane<16: K = k0..k0+15 ; lane>=16: K = k0+16..k0+31  (caller adds half*16).
__device__ __forceinline__ v16bf load_b(const bf16* __restrict__ base, int col,
                                        int k0, int ldk) {
  ABf r;
  const bf16* p = base + (size_t)col * ldk + k0;
  r.q[0] = *(const uint4*)(p);
  r.q[1] = *(const uint4*)(p + 8);
  return r.v;
}

// ---------------------------------------------------------------------------
// Prep: f32 -> bf16 weights + x_0 into workspace, zero spikes + barrier ctr.
// ---------------------------------------------------------------------------
__global__ void bnn_prep(const float* __restrict__ Wr, const float* __restrict__ Wi,
                         const float* __restrict__ Wo, const float* __restrict__ x,
                         bf16* __restrict__ wr_b, bf16* __restrict__ wi_b,
                         bf16* __restrict__ wo_b, bf16* __restrict__ xb0,
                         bf16* __restrict__ s0, unsigned* __restrict__ cnt) {
  size_t i = (size_t)blockIdx.x * blockDim.x + threadIdx.x;
  size_t stride = (size_t)gridDim.x * blockDim.x;
  if (i == 0) *cnt = 0u;
  for (size_t k = i; k < (size_t)NHID * NHID; k += stride) wr_b[k] = (__bf16)Wr[k];
  for (size_t k = i; k < (size_t)NHID * NIN; k += stride)  wi_b[k] = (__bf16)Wi[k];
  for (size_t k = i; k < (size_t)NOUT * NHID; k += stride) wo_b[k] = (__bf16)Wo[k];
  for (size_t k = i; k < (size_t)BSZ * NIN; k += stride)   xb0[k] = (__bf16)x[k];
  for (size_t k = i; k < (size_t)BSZ * NHID; k += stride)  s0[k] = (__bf16)0.0f;
}

// ---------------------------------------------------------------------------
// Persistent recurrent kernel: 16 WGs, one grid barrier per timestep.
// ---------------------------------------------------------------------------
__global__ void __launch_bounds__(BLK, 1) bnn_persistent(
    const float* __restrict__ x,      // [T,B,NIN] f32
    const float* __restrict__ b_in,   // [NHID]
    const float* __restrict__ b_out,  // [NOUT]
    const bf16* __restrict__ wrec_g,  // [NHID,NHID] bf16
    const bf16* __restrict__ win_g,   // [NHID,NIN]  bf16
    const bf16* __restrict__ wout_g,  // [NOUT,NHID] bf16
    bf16* __restrict__ sbuf0, bf16* __restrict__ sbuf1,  // [B,NHID] ping-pong
    bf16* __restrict__ xb0, bf16* __restrict__ xb1,      // [B,NIN] bf16 ping-pong
    unsigned* __restrict__ cnt,
    float* __restrict__ out)          // [T,B,NOUT] f32
{
  extern __shared__ char smem[];
  float* y_s  = (float*)smem;                 // [256][33] padded
  float* v_s  = y_s + 256 * 33;               // [256*32]
  float* a0_s = v_s + 8192;
  float* a1_s = a0_s + 8192;
  float* sp_s = a1_s + 8192;                  // f32 spike state (slice)
  bf16*  wr_s = (bf16*)(sp_s + 8192);         // [32][WREC_LD]
  bf16*  wi_s = wr_s + HSLICE * WREC_LD;      // [32][WIN_LD]

  const int wg   = blockIdx.x;
  const int tid  = threadIdx.x;
  const int wave = tid >> 5;
  const int lane = tid & 31;
  const int lrow = lane & 15;
  const int half = lane >> 4;
  const int h0   = wg * HSLICE;

  // Stage our W_rec / W_in column-slice into LDS, column-major [n][K], padded.
  for (int idx = tid; idx < HSLICE * NHID; idx += BLK) {
    int j = idx >> 9, k = idx & 511;
    wr_s[j * WREC_LD + k] = wrec_g[(size_t)(h0 + j) * NHID + k];
  }
  for (int idx = tid; idx < HSLICE * NIN; idx += BLK) {
    int j = idx >> 7, k = idx & 127;
    wi_s[j * WIN_LD + k] = win_g[(size_t)(h0 + j) * NIN + k];
  }
  for (int i = tid; i < 8192; i += BLK) {
    v_s[i] = 0.0f; a0_s[i] = 0.0f; a1_s[i] = 0.0f; sp_s[i] = 0.0f;
  }
  __syncthreads();

  const int   jcol  = tid & 31;          // hidden col (local) this thread updates
  const int   brow0 = tid >> 5;          // first batch row
  const float binj  = 0.5f * b_in[h0 + jcol];
  const float bo_n  = b_out[wave * 16 + lrow];
  // GLIFR constants (DT=0.05, k_m=0.2, R=0.1+1/512, I0=700, sigma_v=50)
  const float ASC0_DEC = 0.85f, ASC1_DEC = -0.5f, ASC_IN = -0.05f;
  const float V_DEC = 0.99f, V_GAIN = 0.01f * 0.101953125f;

  for (int t = 0; t < T_STEPS; ++t) {
    const bf16* sprev = (t & 1) ? sbuf1 : sbuf0;
    bf16*       snext = (t & 1) ? sbuf0 : sbuf1;
    const bf16* xcur  = (t & 1) ? xb1 : xb0;
    bf16*       xnext = (t & 1) ? xb0 : xb1;

    // Opaque zero: folded into every B-operand index so those loads are not
    // loop-invariant -> compiler cannot hoist 128-VGPR weight tiles across
    // the t-loop (which was starving the scheduler and serializing A loads).
    int tzero = 0;
    asm volatile("" : "+s"(tzero));

    // ---- y[:, h0:h0+32] = 0.5*(x_t @ Win^T + s_prev @ Wrec^T) -------------
#pragma unroll
    for (int mi = 0; mi < 2; ++mi) {
      const int m0 = (wave + mi * 8) * 16;
      const int ar = m0 + lrow;
      v8f acc0 = {}, acc1 = {};

      // input projection: K=128 -> one chunk of 4 K-steps
      {
        v16bf aC[4], b0[4], b1[4];
#pragma unroll
        for (int u = 0; u < 4; ++u) {
          aC[u] = load_a_g(xcur, ar, u * 32 + half * 16, NIN);
          b0[u] = load_b(wi_s, lrow,      u * 32 + half * 16 + tzero, WIN_LD);
          b1[u] = load_b(wi_s, 16 + lrow, u * 32 + half * 16 + tzero, WIN_LD);
        }
#pragma unroll
        for (int u = 0; u < 4; ++u) {
          acc0 = wmma_bf16(aC[u], b0[u], acc0);
          acc1 = wmma_bf16(aC[u], b1[u], acc1);
        }
      }
      // recurrent: K=512 -> 4 chunks of 4 K-steps, A prefetched 1 chunk ahead
      {
        v16bf aC[4], aN[4], b0[4], b1[4];
#pragma unroll
        for (int u = 0; u < 4; ++u)
          aC[u] = load_a_g(sprev, ar, u * 32 + half * 16, NHID);
#pragma unroll
        for (int kt = 0; kt < 4; ++kt) {
          if (kt < 3) {
#pragma unroll
            for (int u = 0; u < 4; ++u)
              aN[u] = load_a_g(sprev, ar, ((kt + 1) * 4 + u) * 32 + half * 16, NHID);
          }
#pragma unroll
          for (int u = 0; u < 4; ++u) {
            int kb = (kt * 4 + u) * 32;
            b0[u] = load_b(wr_s, lrow,      kb + half * 16 + tzero, WREC_LD);
            b1[u] = load_b(wr_s, 16 + lrow, kb + half * 16 + tzero, WREC_LD);
          }
#pragma unroll
          for (int u = 0; u < 4; ++u) {
            acc0 = wmma_bf16(aC[u], b0[u], acc0);
            acc1 = wmma_bf16(aC[u], b1[u], acc1);
          }
          // scheduling: issue next-chunk A first, then (2 DS, 1 WMMA) x 8
          if (kt < 3) SGB(0x020, 8, 0);
#pragma unroll
          for (int g = 0; g < 8; ++g) { SGB(0x100, 2, 0); SGB(0x008, 1, 0); }
          if (kt < 3) {
#pragma unroll
            for (int u = 0; u < 4; ++u) aC[u] = aN[u];
          }
        }
      }
#pragma unroll
      for (int r = 0; r < 8; ++r) {   // D layout: VGPR r -> M = half*8+r, N = lrow
        int m = m0 + half * 8 + r;
        y_s[m * 33 + lrow]      = 0.5f * acc0[r];
        y_s[m * 33 + 16 + lrow] = 0.5f * acc1[r];
      }
    }

    // ---- stage x_{t+1} as bf16 (off critical path; published by barrier) --
    if (t + 1 < T_STEPS) {
      const float* xn = x + (size_t)(t + 1) * BSZ * NIN + wg * 16 * NIN;
      bf16* xd = xnext + (size_t)wg * 16 * NIN;
      const int e = tid * 8;          // 2048 elems per WG, 8 per thread
      float4 f0 = *(const float4*)(xn + e);
      float4 f1 = *(const float4*)(xn + e + 4);
      ABf pk;
      pk.v[0] = (__bf16)f0.x; pk.v[1] = (__bf16)f0.y;
      pk.v[2] = (__bf16)f0.z; pk.v[3] = (__bf16)f0.w;
      pk.v[4] = (__bf16)f1.x; pk.v[5] = (__bf16)f1.y;
      pk.v[6] = (__bf16)f1.z; pk.v[7] = (__bf16)f1.w;
      *(uint4*)(xd + e) = pk.q[0];
      if (t + 2 < T_STEPS)            // warm L2 for the step after next
        __builtin_prefetch(x + (size_t)(t + 2) * BSZ * NIN + wg * 16 * NIN + e, 0, 3);
    }
    __syncthreads();

    // ---- elementwise GLIFR update (state lives in LDS) --------------------
#pragma unroll 4
    for (int i = 0; i < 32; ++i) {
      int bb = brow0 + i * 8;
      int e  = bb * 32 + jcol;
      float y  = y_s[bb * 33 + jcol] + binj;
      float sp = sp_s[e];
      float a0 = a0_s[e] * ASC0_DEC + ASC_IN * sp;
      float a1 = a1_s[e] * ASC1_DEC + ASC_IN * sp;
      float it = y + a0 + a1 + 700.0f;
      float v  = v_s[e] * V_DEC * (1.0f - 0.05f * sp) + V_GAIN * it;
      float sn = 20.0f / (1.0f + __expf(v * -0.02f));
      v_s[e] = v; a0_s[e] = a0; a1_s[e] = a1; sp_s[e] = sn;
      snext[(size_t)bb * NHID + h0 + jcol] = (__bf16)sn;
    }

    // ---- publish s_new + x_{t+1}, grid barrier ----------------------------
    __threadfence();
    __syncthreads();
    if (tid == 0) {
      __hip_atomic_fetch_add(cnt, 1u, __ATOMIC_ACQ_REL, __HIP_MEMORY_SCOPE_AGENT);
      const unsigned target = (unsigned)(t + 1) * NWG;
      while (__hip_atomic_load(cnt, __ATOMIC_ACQUIRE, __HIP_MEMORY_SCOPE_AGENT) <
             target)
        __builtin_amdgcn_s_sleep(1);
    }
    __syncthreads();
    __threadfence();
    __builtin_amdgcn_s_cluster_barrier();  // NOP unless dispatched as a cluster

    // ---- readout: out[t, wg*16:+16, :] = s_new @ Wout^T + b_out -----------
    // Both operands stream from L2 -> double-buffer A and B one chunk ahead.
    {
      const int m0r = wg * 16;
      const int n0  = wave * 16;
      v8f acc = {};
      v16bf aC[4], aN[4], bC[4], bN[4];
#pragma unroll
      for (int u = 0; u < 4; ++u) {
        aC[u] = load_a_g(snext, m0r + lrow, u * 32 + half * 16, NHID);
        bC[u] = load_b(wout_g, n0 + lrow, u * 32 + half * 16 + tzero, NHID);
      }
#pragma unroll
      for (int kt = 0; kt < 4; ++kt) {
        if (kt < 3) {
#pragma unroll
          for (int u = 0; u < 4; ++u) {
            int kb = ((kt + 1) * 4 + u) * 32;
            aN[u] = load_a_g(snext, m0r + lrow, kb + half * 16, NHID);
            bN[u] = load_b(wout_g, n0 + lrow, kb + half * 16 + tzero, NHID);
          }
        }
#pragma unroll
        for (int u = 0; u < 4; ++u) acc = wmma_bf16(aC[u], bC[u], acc);
        if (kt < 3) SGB(0x020, 16, 0);   // next-chunk A+B loads first
        SGB(0x008, 4, 0);                // then this chunk's 4 WMMA
        if (kt < 3) {
#pragma unroll
          for (int u = 0; u < 4; ++u) { aC[u] = aN[u]; bC[u] = bN[u]; }
        }
      }
      float* ot = out + (size_t)t * BSZ * NOUT;
#pragma unroll
      for (int r = 0; r < 8; ++r) {
        int m = m0r + half * 8 + r;
        ot[(size_t)m * NOUT + n0 + lrow] = acc[r] + bo_n;
      }
    }
  }
}

extern "C" void kernel_launch(void* const* d_in, const int* in_sizes, int n_in,
                              void* d_out, int out_size, void* d_ws, size_t ws_size,
                              hipStream_t stream) {
  (void)in_sizes; (void)n_in; (void)out_size; (void)ws_size;
  const float* x  = (const float*)d_in[0];
  const float* Wi = (const float*)d_in[1];
  const float* bi = (const float*)d_in[2];
  const float* Wr = (const float*)d_in[3];
  const float* Wo = (const float*)d_in[4];
  const float* bo = (const float*)d_in[5];
  float* out = (float*)d_out;

  // Workspace (~1.5 MB): ping-pong spikes + x, bf16 weights, barrier ctr.
  bf16* s0   = (bf16*)d_ws;
  bf16* s1   = s0 + (size_t)BSZ * NHID;
  bf16* wr_b = s1 + (size_t)BSZ * NHID;
  bf16* wi_b = wr_b + (size_t)NHID * NHID;
  bf16* wo_b = wi_b + (size_t)NHID * NIN;
  bf16* xb0  = wo_b + (size_t)NOUT * NHID;
  bf16* xb1  = xb0 + (size_t)BSZ * NIN;
  unsigned* cnt = (unsigned*)(xb1 + (size_t)BSZ * NIN);

  bnn_prep<<<512, 256, 0, stream>>>(Wr, Wi, Wo, x, wr_b, wi_b, wo_b, xb0, s0, cnt);

  const size_t smem = (256 * 33 + 4 * 8192) * sizeof(float) +
                      (HSLICE * WREC_LD + HSLICE * WIN_LD) * sizeof(bf16);
  hipFuncSetAttribute((const void*)bnn_persistent,
                      hipFuncAttributeMaxDynamicSharedMemorySize, (int)smem);
  bnn_persistent<<<NWG, BLK, smem, stream>>>(x, bi, bo, wr_b, wi_b, wo_b, s0, s1,
                                             xb0, xb1, cnt, out);
}
